// OCPolicy_11355893530644
// MI455X (gfx1250) — compile-verified, matrix operand order checked by reading, not run on hardware
//
#include <hip/hip_runtime.h>
#include <hip/hip_bf16.h>

typedef _Float16 half_t;
typedef __attribute__((ext_vector_type(16))) _Float16 v16h;
typedef __attribute__((ext_vector_type(8)))  _Float16 v8h;
typedef __attribute__((ext_vector_type(8)))  float    v8f;

// LDS staging geometry (per workgroup): 64 rows of activations.
#define ASTRIDE 640            // f16 A-buffer column stride (max K = 640)
#define CSTRIDE 512            // f32 C-buffer column stride (max Nout = 512)
#define LDS_C_BYTES (64 * CSTRIDE * 4)   // 131072
#define LDS_A_BYTES (64 * ASTRIDE * 2)   // 81920
#define LDS_BYTES   (LDS_C_BYTES + LDS_A_BYTES)  // 212992 <= 320KB WGP LDS

// ---- d_ws layout (bytes) ----
#define WS_EW1H 0u               // 512*256 halves
#define WS_EW2H 262144u          // 512*512 halves
#define WS_EW3H 786432u          // 512*512 halves
#define WS_NW1H 1310720u         // 512*640 halves
#define WS_NW2H 1966080u         // 512*512 halves
#define WS_NW3H 2490368u         // 128*512 halves
#define WS_AGG  2621440u         // B*8*512 floats

// Load a 16-half A/B fragment for v_wmma_f32_16x16x32_f16.
// Per ISA 7.12.2 (16-bit A 16x32): lane's elements 0..7 are K=kb..kb+7,
// elements 8..15 are K=kb+16..kb+23, where kb already includes the +8 for
// lanes 16..31. Two aligned 16-byte loads.
__device__ __forceinline__ v16h frag16(const half_t* p) {
  v8h lo = *(const v8h*)(p);
  v8h hi = *(const v8h*)(p + 16);
  v16h r;
#pragma unroll
  for (int i = 0; i < 8; ++i) { r[i] = lo[i]; r[i + 8] = hi[i]; }
  return r;
}

// One fused GEMM stage: actC[64xNOUT] = actA[64xK](f16) * W^T(f16) + bias.
// W is (NOUT x K) row-major f16 (so B[k][n] = W[n][k], loaded column-major
// with the same per-lane striping as A).
//
// Blocking: each of the 8 waves exclusively owns NTILES/8 N-tiles and walks
// ALL 4 M-tiles with 4 live accumulators, so every weight fragment is
// fetched from L2 exactly once per workgroup; A-fragment reuse traffic lands
// on the per-WGP LDS. The k-loop is software-pipelined: fragments for k+1
// (1 global B pair + 4 LDS A pairs) are issued before the current 4 WMMAs so
// DS/L2 latency is covered by a full WMMA quad instead of exposed in
// s_wait_dscnt 0 stalls (only ~2 waves/SIMD at this LDS footprint).
// C/D layout: col = lane%16, row = vgpr + (lane>=16 ? 8:0).
template <int K, int NOUT>
__device__ __forceinline__ void gemm_stage(const half_t* actA, float* actC,
                                           const half_t* W, const float* bias) {
  const int tid   = (int)threadIdx.x;
  const int w     = tid >> 5;
  const int lane  = tid & 31;
  const int lrow  = lane & 15;
  const int khi   = (lane & 16) ? 8 : 0;
  const int hi8   = (lane & 16) ? 8 : 0;

  constexpr int NTILES = NOUT >> 4;
  constexpr int NPW    = NTILES >> 3;     // n-tiles per wave (4 or 1)
  constexpr int KC     = K >> 5;          // k-chunks of 32
  const int nstart = w * NPW;

  // A-fragment row pointers for the 4 M-tiles (rows m = mt*16 + lrow).
  const half_t* arow0 = actA + (0 * 16 + lrow) * ASTRIDE;
  const half_t* arow1 = actA + (1 * 16 + lrow) * ASTRIDE;
  const half_t* arow2 = actA + (2 * 16 + lrow) * ASTRIDE;
  const half_t* arow3 = actA + (3 * 16 + lrow) * ASTRIDE;

#pragma unroll 1
  for (int nt = 0; nt < NPW; ++nt) {
    const int ntile = nstart + nt;
    const int n = ntile * 16 + lrow;
    const half_t* wrow = W + (size_t)n * K;

    v8f c0 = {0.f, 0.f, 0.f, 0.f, 0.f, 0.f, 0.f, 0.f};
    v8f c1 = c0, c2 = c0, c3 = c0;

    // Pipeline prologue: fragments for kk = 0.
    v16h b  = frag16(wrow  + khi);
    v16h a0 = frag16(arow0 + khi);
    v16h a1 = frag16(arow1 + khi);
    v16h a2 = frag16(arow2 + khi);
    v16h a3 = frag16(arow3 + khi);

#pragma unroll
    for (int kk = 0; kk < KC; ++kk) {
      v16h bn, an0, an1, an2, an3;
      if (kk + 1 < KC) {            // prefetch next k-chunk before computing
        const int kb = (kk + 1) * 32 + khi;
        bn  = frag16(wrow  + kb);
        an0 = frag16(arow0 + kb);
        an1 = frag16(arow1 + kb);
        an2 = frag16(arow2 + kb);
        an3 = frag16(arow3 + kb);
      }
      c0 = __builtin_amdgcn_wmma_f32_16x16x32_f16(false, a0, false, b, (short)0, c0, false, false);
      c1 = __builtin_amdgcn_wmma_f32_16x16x32_f16(false, a1, false, b, (short)0, c1, false, false);
      c2 = __builtin_amdgcn_wmma_f32_16x16x32_f16(false, a2, false, b, (short)0, c2, false, false);
      c3 = __builtin_amdgcn_wmma_f32_16x16x32_f16(false, a3, false, b, (short)0, c3, false, false);
      if (kk + 1 < KC) {
        b = bn; a0 = an0; a1 = an1; a2 = an2; a3 = an3;
      }
    }

    const float bv = bias[n];
#pragma unroll
    for (int v = 0; v < 8; ++v) {
      actC[(0 * 16 + hi8 + v) * CSTRIDE + n] = c0[v] + bv;
      actC[(1 * 16 + hi8 + v) * CSTRIDE + n] = c1[v] + bv;
      actC[(2 * 16 + hi8 + v) * CSTRIDE + n] = c2[v] + bv;
      actC[(3 * 16 + hi8 + v) * CSTRIDE + n] = c3[v] + bv;
    }
  }
}

// actA = f16(relu(actC)), first NCOLS columns.
template <int NCOLS>
__device__ __forceinline__ void stage_relu(const float* actC, half_t* actA) {
  for (int idx = (int)threadIdx.x; idx < 64 * NCOLS; idx += 256) {
    const int r = idx / NCOLS, c = idx % NCOLS;
    const float v = actC[r * CSTRIDE + c];
    actA[r * ASTRIDE + c] = (half_t)(v > 0.f ? v : 0.f);
  }
}

// actA = f16(relu(layernorm(actC) * g + b)) over 512 columns.
__device__ __forceinline__ void stage_ln_relu(const float* actC, half_t* actA,
                                              const float* g, const float* bb,
                                              float* rs_m, float* rs_r) {
  const int tid = (int)threadIdx.x;
  if (tid < 64) {
    float s = 0.f, s2 = 0.f;
    for (int c = 0; c < 512; ++c) {
      const float v = actC[tid * CSTRIDE + c];
      s += v; s2 += v * v;
    }
    const float m   = s * (1.f / 512.f);
    const float var = s2 * (1.f / 512.f) - m * m;
    rs_m[tid] = m;
    rs_r[tid] = rsqrtf(var + 1e-5f);
  }
  __syncthreads();
  for (int idx = tid; idx < 64 * 512; idx += 256) {
    const int r = idx >> 9, c = idx & 511;
    const float v = (actC[r * CSTRIDE + c] - rs_m[r]) * rs_r[r] * g[c] + bb[c];
    actA[r * ASTRIDE + c] = (half_t)(v > 0.f ? v : 0.f);
  }
}

__global__ void cvt_f32_to_f16(const float* __restrict__ src,
                               half_t* __restrict__ dst, int n) {
  const int i = blockIdx.x * 256 + (int)threadIdx.x;
  if (i < n) dst[i] = (half_t)src[i];
}

// ---- Edge MLP + aggregation: one workgroup per batch (M = 64 = 56 edges pad)
__global__ void edge_kernel(const float* __restrict__ slots,
                            const half_t* __restrict__ ew1h, const float* __restrict__ eb1,
                            const half_t* __restrict__ ew2h, const float* __restrict__ eb2,
                            const float* __restrict__ eln_g, const float* __restrict__ eln_b,
                            const half_t* __restrict__ ew3h, const float* __restrict__ eb3,
                            float* __restrict__ agg) {
  extern __shared__ __align__(16) char smem[];
  float*  actC = (float*)smem;
  half_t* actA = (half_t*)(smem + LDS_C_BYTES);
  __shared__ float rs_m[64], rs_r[64];

  const int b = blockIdx.x;
  const float* sb = slots + (size_t)b * 8 * 128;

  // Build edge features: row e -> [slots[ROW[e]] | slots[COL[e]]], rows>=56 zero.
  for (int idx = (int)threadIdx.x; idx < 64 * 256; idx += 256) {
    const int e = idx >> 8, c = idx & 255;
    float v = 0.f;
    if (e < 56) {
      const int r = e / 7, j = e % 7;
      const int node = (c < 128) ? r : (j < r ? j : j + 1);
      v = sb[node * 128 + (c & 127)];
    }
    actA[e * ASTRIDE + c] = (half_t)v;
  }
  __syncthreads();

  gemm_stage<256, 512>(actA, actC, ew1h, eb1);
  __syncthreads();
  stage_relu<512>(actC, actA);
  __syncthreads();

  gemm_stage<512, 512>(actA, actC, ew2h, eb2);
  __syncthreads();
  stage_ln_relu(actC, actA, eln_g, eln_b, rs_m, rs_r);
  __syncthreads();

  gemm_stage<512, 512>(actA, actC, ew3h, eb3);
  __syncthreads();

  // Scatter-add: node n receives edges e = n*7 .. n*7+6 (ROW[e] == n).
  for (int idx = (int)threadIdx.x; idx < 8 * 512; idx += 256) {
    const int n = idx >> 9, c = idx & 511;
    float s = 0.f;
#pragma unroll
    for (int j = 0; j < 7; ++j) s += actC[(n * 7 + j) * CSTRIDE + c];
    agg[((size_t)b * 8 + n) * 512 + c] = s;
  }
}

// ---- Node MLP + pool + head: one workgroup per 8 batches (M = 64 rows)
__global__ void node_kernel(const float* __restrict__ slots,
                            const float* __restrict__ agg,
                            const half_t* __restrict__ nw1h, const float* __restrict__ nb1,
                            const half_t* __restrict__ nw2h, const float* __restrict__ nb2,
                            const float* __restrict__ nln_g, const float* __restrict__ nln_b,
                            const half_t* __restrict__ nw3h, const float* __restrict__ nb3,
                            const float* __restrict__ head_w, const float* __restrict__ head_b,
                            float* __restrict__ out) {
  extern __shared__ __align__(16) char smem[];
  float*  actC = (float*)smem;
  half_t* actA = (half_t*)(smem + LDS_C_BYTES);
  __shared__ float rs_m[64], rs_r[64];
  __shared__ float pooled[8 * 128];

  const size_t b0 = (size_t)blockIdx.x * 8;

  // Build node features: row r = (local batch)*8 + node -> [slots | agg].
  for (int idx = (int)threadIdx.x; idx < 64 * 640; idx += 256) {
    const int r = idx / 640, c = idx % 640;
    const int bi = r >> 3, n = r & 7;
    float v;
    if (c < 128) v = slots[((b0 + bi) * 8 + n) * 128 + c];
    else         v = agg[((b0 + bi) * 8 + n) * 512 + (c - 128)];
    actA[r * ASTRIDE + c] = (half_t)v;
  }
  __syncthreads();

  gemm_stage<640, 512>(actA, actC, nw1h, nb1);
  __syncthreads();
  stage_relu<512>(actC, actA);
  __syncthreads();

  gemm_stage<512, 512>(actA, actC, nw2h, nb2);
  __syncthreads();
  stage_ln_relu(actC, actA, nln_g, nln_b, rs_m, rs_r);
  __syncthreads();

  gemm_stage<512, 128>(actA, actC, nw3h, nb3);
  __syncthreads();

  // relu + pool over 8 nodes.
  for (int idx = (int)threadIdx.x; idx < 8 * 128; idx += 256) {
    const int bi = idx >> 7, c = idx & 127;
    float s = 0.f;
#pragma unroll
    for (int n = 0; n < 8; ++n) {
      const float v = actC[(bi * 8 + n) * CSTRIDE + c];
      s += (v > 0.f ? v : 0.f);
    }
    pooled[bi * 128 + c] = s;
  }
  __syncthreads();

  // Head: (8 x 128) @ (12 x 128)^T + head_b  -> tiny, VALU is fine.
  if (threadIdx.x < 96) {
    const int bi = (int)threadIdx.x / 12, o = (int)threadIdx.x % 12;
    float s = head_b[o];
    for (int c = 0; c < 128; ++c) s += pooled[bi * 128 + c] * head_w[o * 128 + c];
    out[(b0 + bi) * 12 + o] = s;
  }
}

extern "C" void kernel_launch(void* const* d_in, const int* in_sizes, int n_in,
                              void* d_out, int out_size, void* d_ws, size_t ws_size,
                              hipStream_t stream) {
  const float* slots  = (const float*)d_in[0];
  const float* ew1    = (const float*)d_in[1];
  const float* eb1    = (const float*)d_in[2];
  const float* ew2    = (const float*)d_in[3];
  const float* eb2    = (const float*)d_in[4];
  const float* eln_g  = (const float*)d_in[5];
  const float* eln_b  = (const float*)d_in[6];
  const float* ew3    = (const float*)d_in[7];
  const float* eb3    = (const float*)d_in[8];
  const float* nw1    = (const float*)d_in[9];
  const float* nb1    = (const float*)d_in[10];
  const float* nw2    = (const float*)d_in[11];
  const float* nb2    = (const float*)d_in[12];
  const float* nln_g  = (const float*)d_in[13];
  const float* nln_b  = (const float*)d_in[14];
  const float* nw3    = (const float*)d_in[15];
  const float* nb3    = (const float*)d_in[16];
  const float* head_w = (const float*)d_in[17];
  const float* head_b = (const float*)d_in[18];
  float* out = (float*)d_out;

  char* ws = (char*)d_ws;
  half_t* ew1h = (half_t*)(ws + WS_EW1H);
  half_t* ew2h = (half_t*)(ws + WS_EW2H);
  half_t* ew3h = (half_t*)(ws + WS_EW3H);
  half_t* nw1h = (half_t*)(ws + WS_NW1H);
  half_t* nw2h = (half_t*)(ws + WS_NW2H);
  half_t* nw3h = (half_t*)(ws + WS_NW3H);
  float*  agg  = (float*)(ws + WS_AGG);

  const int B = in_sizes[0] / (8 * 128);   // 2048

  cvt_f32_to_f16<<<(131072 + 255) / 256, 256, 0, stream>>>(ew1, ew1h, 131072);
  cvt_f32_to_f16<<<(262144 + 255) / 256, 256, 0, stream>>>(ew2, ew2h, 262144);
  cvt_f32_to_f16<<<(262144 + 255) / 256, 256, 0, stream>>>(ew3, ew3h, 262144);
  cvt_f32_to_f16<<<(327680 + 255) / 256, 256, 0, stream>>>(nw1, nw1h, 327680);
  cvt_f32_to_f16<<<(262144 + 255) / 256, 256, 0, stream>>>(nw2, nw2h, 262144);
  cvt_f32_to_f16<<<(65536  + 255) / 256, 256, 0, stream>>>(nw3, nw3h, 65536);

  (void)hipFuncSetAttribute((const void*)edge_kernel,
                            hipFuncAttributeMaxDynamicSharedMemorySize, LDS_BYTES);
  (void)hipFuncSetAttribute((const void*)node_kernel,
                            hipFuncAttributeMaxDynamicSharedMemorySize, LDS_BYTES);

  edge_kernel<<<B, 256, LDS_BYTES, stream>>>(slots, ew1h, eb1, ew2h, eb2,
                                             eln_g, eln_b, ew3h, eb3, agg);
  node_kernel<<<B / 8, 256, LDS_BYTES, stream>>>(slots, agg, nw1h, nb1, nw2h, nb2,
                                                 nln_g, nln_b, nw3h, nb3,
                                                 head_w, head_b, out);
}
